// HeteroClassifier_87308095193388
// MI455X (gfx1250) — matrix operand order repaired and checked by reading, not compiled, not run on hardware
//
#include <hip/hip_runtime.h>

typedef __attribute__((ext_vector_type(16))) _Float16 v16h;
typedef __attribute__((ext_vector_type(8)))  float    v8f;

#define NN   100000      // real nodes
#define NPAD 100096      // padded nodes = 782*128
#define NE   400000      // edges per relation
#define NF   256         // features
#define NG   128         // graphs
#define NREL 3

// ---------------------------------------------------------------- utilities
__global__ void k_zero(float* __restrict__ p, long n) {
  long i = (long)blockIdx.x * blockDim.x + threadIdx.x;
  long stride = (long)gridDim.x * blockDim.x;
  for (; i < n; i += stride) p[i] = 0.f;
}

// degrees for all 3 relations at once (edge arrays are [3][NE] contiguous)
__global__ void k_degrees(const int* __restrict__ src, const int* __restrict__ dst,
                          float* __restrict__ deg_out, float* __restrict__ deg_in) {
  long i = (long)blockIdx.x * blockDim.x + threadIdx.x;
  if (i >= (long)NREL * NE) return;
  int r = (int)(i / NE);
  atomicAdd(&deg_out[(size_t)r * NPAD + src[i]], 1.f);
  atomicAdd(&deg_in [(size_t)r * NPAD + dst[i]], 1.f);
}

// deg -> (deg>0 ? 1/sqrt(deg) : 0), in place (padding rows stay 0)
__global__ void k_norm(float* __restrict__ deg, long n) {
  long i = (long)blockIdx.x * blockDim.x + threadIdx.x;
  if (i >= n) return;
  float d = deg[i];
  deg[i] = (d > 0.f) ? rsqrtf(d) : 0.f;
}

// acc[i][j] = b[0][j] + b[1][j] + b[2][j]   (bias summed over relations)
__global__ void k_init_acc(float* __restrict__ acc, const float* __restrict__ b) {
  long i = (long)blockIdx.x * blockDim.x + threadIdx.x;
  if (i >= (long)NPAD * NF) return;
  int j = (int)(i & (NF - 1));
  acc[i] = b[j] + b[NF + j] + b[2 * NF + j];
}

__global__ void k_relu(const float* __restrict__ a, float* __restrict__ h, long n) {
  long i = (long)blockIdx.x * blockDim.x + threadIdx.x;
  if (i >= n) return;
  h[i] = fmaxf(a[i], 0.f);
}

// ------------------------------------------------------------------- SpMM
// agg[d[e]] += x[s[e]] * norm_src[s[e]] ; 64 threads per edge, float4 chunks
__global__ void k_spmm(const float* __restrict__ x, const int* __restrict__ s,
                       const int* __restrict__ d, const float* __restrict__ ns,
                       float* __restrict__ agg) {
  long t = (long)blockIdx.x * blockDim.x + threadIdx.x;
  if (t >= (long)NE * 64) return;
  int  c = (int)(t & 63);
  long e = t >> 6;
  int si = s[e], di = d[e];
  float nrm = ns[si];
  const float4 v = *(const float4*)(x + (size_t)si * NF + c * 4);
  float* o = agg + (size_t)di * NF + c * 4;
  atomicAdd(o + 0, v.x * nrm);
  atomicAdd(o + 1, v.y * nrm);
  atomicAdd(o + 2, v.z * nrm);
  atomicAdd(o + 3, v.w * nrm);
}

// ----------------------------------------------------- WMMA fragment packing
// A-fragment (16x32 f16): lane L -> row m=L&15, kb=(L>>4)*8
//   halfs 0..7  = A[m][kt*32+kb .. +kb+7]
//   halfs 8..15 = A[m][kt*32+16+kb .. ]
__global__ void k_pack_a(const float* __restrict__ A, _Float16* __restrict__ Ap) {
  long t = (long)blockIdx.x * blockDim.x + threadIdx.x;   // (NPAD/16)*8*32 threads
  if (t >= (long)(NPAD / 16) * 8 * 32) return;
  int lane = (int)(t & 31);
  int kt   = (int)((t >> 5) & 7);
  long mt  = t >> 8;
  int m  = (int)mt * 16 + (lane & 15);
  int c0 = kt * 32 + (lane >> 4) * 8;
  const float* row = A + (size_t)m * NF;
  _Float16* out = Ap + (((size_t)mt * 8 + kt) * 32 + lane) * 16;
#pragma unroll
  for (int j = 0; j < 8; ++j) out[j] = (_Float16)row[c0 + j];
#pragma unroll
  for (int j = 0; j < 8; ++j) out[8 + j] = (_Float16)row[c0 + 16 + j];
}

// B-fragment (32x16 f16): lane L -> col n=L&15, kb=(L>>4)*8
//   halfs t=0..7 = W[kt*32+kb+t][nt*16+n], t=8..15 = W[kt*32+16+kb+t-8][..]   (packs one layer's 3 relations)
__global__ void k_pack_w(const float* __restrict__ W, _Float16* __restrict__ Wp) {
  int t = blockIdx.x * blockDim.x + threadIdx.x;          // NREL*8*16*32 threads
  if (t >= NREL * 8 * 16 * 32) return;
  int lane = t & 31;
  int nt = (t >> 5) & 15;
  int kt = (t >> 9) & 7;
  int r  = t >> 12;
  int col = nt * 16 + (lane & 15);
  int k0  = kt * 32 + (lane >> 4) * 8;
  const float* Wr = W + (size_t)r * NF * NF;
  _Float16* out = Wp + (size_t)r * 65536 + (((size_t)kt * 16 + nt) * 32 + lane) * 16;
#pragma unroll
  for (int tt = 0; tt < 16; ++tt) out[tt] = (_Float16)Wr[(size_t)(k0 + ((tt & 8) << 1) + (tt & 7)) * NF + col];
}

// ------------------------------------------------------------------- GEMM
// acc[m][:] += (Ap_tile @ Wp) * norm_dst[m]
// block = 8 waves; wave w -> m-tile = blockIdx.x*8+w (16 rows)
// blockIdx.y selects cols [y*128, y*128+128); wave holds 8 v8f accumulators.
__global__ __launch_bounds__(256)
void k_gemm(const _Float16* __restrict__ Ap, const _Float16* __restrict__ Wp,
            const float* __restrict__ nd, float* __restrict__ acc) {
  const int wave = threadIdx.x >> 5;
  const int lane = threadIdx.x & 31;
  const long mt = (long)blockIdx.x * 8 + wave;
  const int nbase = blockIdx.y * 8;                       // n-tile base (of 16)

  v8f C[8];
#pragma unroll
  for (int i = 0; i < 8; ++i) C[i] = (v8f){0.f, 0.f, 0.f, 0.f, 0.f, 0.f, 0.f, 0.f};

#pragma unroll
  for (int kt = 0; kt < 8; ++kt) {
    v16h a = *(const v16h*)(Ap + (((size_t)mt * 8 + kt) * 32 + lane) * 16);
#pragma unroll
    for (int nt = 0; nt < 8; ++nt) {
      v16h b = *(const v16h*)(Wp + (((size_t)kt * 16 + (nbase + nt)) * 32 + lane) * 16);
      C[nt] = __builtin_amdgcn_wmma_f32_16x16x32_f16(
          false, a, false, b, (short)0, C[nt], false, false);
    }
  }

  const int n0 = lane & 15;
  const int mb = (lane >> 4) * 8;
  const long row0 = mt * 16 + mb;
#pragma unroll
  for (int j = 0; j < 8; ++j) {
    long row = row0 + j;
    float s = nd[row];
    float* o = acc + row * (long)NF + blockIdx.y * 128 + n0;
#pragma unroll
    for (int nt = 0; nt < 8; ++nt)
      o[nt * 16] += C[nt][j] * s;
  }
}

// ------------------------------------------------------------------ pooling
__global__ void k_pool_sum(const float* __restrict__ h, const int* __restrict__ gid,
                           float* __restrict__ sums, float* __restrict__ cnt) {
  int node = blockIdx.x;            // < NN
  int j = threadIdx.x;              // < NF
  int g = gid[node];
  atomicAdd(&sums[(size_t)g * NF + j], h[(size_t)node * NF + j]);
  if (j == 0) atomicAdd(&cnt[g], 1.f);
}

__global__ void k_pool_div(float* __restrict__ out, const float* __restrict__ cnt) {
  int g = blockIdx.x, j = threadIdx.x;
  out[(size_t)g * NF + j] /= fmaxf(cnt[g], 1.f);
}

// ------------------------------------------------------------------- driver
extern "C" void kernel_launch(void* const* d_in, const int* in_sizes, int n_in,
                              void* d_out, int out_size, void* d_ws, size_t ws_size,
                              hipStream_t stream) {
  (void)in_sizes; (void)n_in; (void)out_size; (void)ws_size;
  const float* feat = (const float*)d_in[0];
  const int*   src  = (const int*)d_in[1];
  const int*   dst  = (const int*)d_in[2];
  const int*   gid  = (const int*)d_in[3];
  const float* W1   = (const float*)d_in[4];
  const float* b1   = (const float*)d_in[5];
  const float* W2   = (const float*)d_in[6];
  const float* b2   = (const float*)d_in[7];
  float* out = (float*)d_out;

  // workspace carve-up (all 256B aligned)
  float* deg_out = (float*)d_ws;                       // [3][NPAD] -> norm_src
  float* deg_in  = deg_out + (size_t)3 * NPAD;         // [3][NPAD] -> norm_dst
  float* acc     = deg_in  + (size_t)3 * NPAD;         // [NPAD][NF]
  float* h1      = acc + (size_t)NPAD * NF;            // [NPAD][NF]
  float* agg     = h1  + (size_t)NPAD * NF;            // [NPAD][NF]
  _Float16* Apack = (_Float16*)(agg + (size_t)NPAD * NF);     // NPAD*NF halfs
  _Float16* Wpack = Apack + (size_t)NPAD * NF;                // 6*NF*NF halfs
  float* cnt = (float*)(Wpack + (size_t)6 * NF * NF);         // [NG]

  const long nfeatP = (long)NPAD * NF;
  const dim3 gemm_grid(NPAD / 128, 2);

  // degrees + norms (shared by both layers)
  k_zero<<<(6 * NPAD + 255) / 256, 256, 0, stream>>>(deg_out, (long)6 * NPAD);
  k_degrees<<<((long)NREL * NE + 255) / 256, 256, 0, stream>>>(src, dst, deg_out, deg_in);
  k_norm<<<(6 * NPAD + 255) / 256, 256, 0, stream>>>(deg_out, (long)6 * NPAD);

  // pack weights for both layers
  k_pack_w<<<(NREL * 8 * 16 * 32 + 255) / 256, 256, 0, stream>>>(W1, Wpack);
  k_pack_w<<<(NREL * 8 * 16 * 32 + 255) / 256, 256, 0, stream>>>(W2, Wpack + (size_t)3 * 65536);

  // ---- layer 1 ----
  k_init_acc<<<(nfeatP + 255) / 256, 256, 0, stream>>>(acc, b1);
  for (int r = 0; r < NREL; ++r) {
    k_zero<<<(nfeatP + 255) / 256, 256, 0, stream>>>(agg, nfeatP);
    k_spmm<<<((long)NE * 64 + 255) / 256, 256, 0, stream>>>(
        feat, src + (size_t)r * NE, dst + (size_t)r * NE, deg_out + (size_t)r * NPAD, agg);
    k_pack_a<<<((long)(NPAD / 16) * 8 * 32 + 255) / 256, 256, 0, stream>>>(agg, Apack);
    k_gemm<<<gemm_grid, 256, 0, stream>>>(
        Apack, Wpack + (size_t)r * 65536, deg_in + (size_t)r * NPAD, acc);
  }
  k_relu<<<(nfeatP + 255) / 256, 256, 0, stream>>>(acc, h1, nfeatP);

  // ---- layer 2 ----
  k_init_acc<<<(nfeatP + 255) / 256, 256, 0, stream>>>(acc, b2);
  for (int r = 0; r < NREL; ++r) {
    k_zero<<<(nfeatP + 255) / 256, 256, 0, stream>>>(agg, nfeatP);
    k_spmm<<<((long)NE * 64 + 255) / 256, 256, 0, stream>>>(
        h1, src + (size_t)r * NE, dst + (size_t)r * NE, deg_out + (size_t)r * NPAD, agg);
    k_pack_a<<<((long)(NPAD / 16) * 8 * 32 + 255) / 256, 256, 0, stream>>>(agg, Apack);
    k_gemm<<<gemm_grid, 256, 0, stream>>>(
        Apack, Wpack + (size_t)(3 + r) * 65536, deg_in + (size_t)r * NPAD, acc);
  }

  // ---- mean pooling over graphs ----
  k_zero<<<((long)NG * NF + 255) / 256, 256, 0, stream>>>(out, (long)NG * NF);
  k_zero<<<1, 256, 0, stream>>>(cnt, (long)NG);
  k_pool_sum<<<NN, NF, 0, stream>>>(acc, gid, out, cnt);
  k_pool_div<<<NG, NF, 0, stream>>>(out, cnt);
}
